// GNN_82059645157818
// MI455X (gfx1250) — matrix operand order repaired and checked by reading, not compiled, not run on hardware
//
#include <hip/hip_runtime.h>
#include <cstdint>

// ---------- types for WMMA ----------
typedef __bf16 bf16;
typedef __attribute__((ext_vector_type(16))) __bf16 v16bf;
typedef __attribute__((ext_vector_type(8)))  float  v8f;
typedef __attribute__((ext_vector_type(4)))  int    v4i;

union V16 { v16bf v; v4i q[2]; };

__device__ __forceinline__ float bf2f(bf16 x) {
    unsigned short u = __builtin_bit_cast(unsigned short, x);
    unsigned int w = ((unsigned int)u) << 16;
    return __builtin_bit_cast(float, w);
}
__device__ __forceinline__ bf16 f2bf(float f) {
    unsigned int u = __builtin_bit_cast(unsigned int, f);
    u += 0x7FFFu + ((u >> 16) & 1u);          // round-to-nearest-even
    unsigned short h = (unsigned short)(u >> 16);
    return __builtin_bit_cast(bf16, h);
}

// gfx1250 async global->LDS 16B copy (ASYNCcnt-tracked), ISA 15.18.3 op 98.
__device__ __forceinline__ void async_ld16(unsigned lds_addr, const void* gaddr) {
    asm volatile("global_load_async_to_lds_b128 %0, %1, off"
                 :: "v"(lds_addr), "v"((unsigned long long)(uintptr_t)gaddr)
                 : "memory");
}
__device__ __forceinline__ void wait_async0() {
#if __has_builtin(__builtin_amdgcn_s_wait_asynccnt)
    __builtin_amdgcn_s_wait_asynccnt(0);
#else
    asm volatile("s_wait_asynccnt 0x0" ::: "memory");
#endif
}

// ---------- problem constants ----------
#define NFc 133
#define EFc 14
#define Hc  300
#define Gc  200
#define Tc  12
#define Bc  4096
#define Nc  131072   // B * 32 nodes
#define Ec  262144   // B * 32 * 2 edges
#define R1c 550
#define R2c 378
// padded dims: K multiples of 32, N multiples of 64 (block tile BN=64)
#define KIp 160      // 147 -> 160
#define HP  320      // 300 -> 320
#define KAp 448      // 433 -> 448
#define KRp 512      // 500 -> 512
#define R1p 576      // 550 -> 576
#define R2p 384      // 378 -> 384
#define TP  64       // 12  -> 64 (shares the tiled GEMM; only cols<12 stored)

// ---------- elementwise kernels ----------
__global__ void k_zero_f32(float* p, int n) {
    int i = blockIdx.x * blockDim.x + threadIdx.x;
    if (i < n) p[i] = 0.f;
}

__global__ void k_convw(const float* __restrict__ W, bf16* __restrict__ o,
                        int Kreal, int Nreal, int Kpad, int Npad) {
    int i = blockIdx.x * blockDim.x + threadIdx.x;
    int n = Kpad * Npad;
    if (i >= n) return;
    int k = i / Npad, c = i % Npad;
    float x = (k < Kreal && c < Nreal) ? W[(size_t)k * Nreal + c] : 0.f;
    o[i] = f2bf(x);
}

__global__ void k_build_xe(const float* __restrict__ v, const float* __restrict__ ef,
                           const int* __restrict__ src, bf16* __restrict__ Xe) {
    int i = blockIdx.x * blockDim.x + threadIdx.x;
    if (i >= Ec * KIp) return;
    int e = i / KIp, c = i % KIp;
    float x = 0.f;
    if (c < NFc)            x = v[(size_t)src[e] * (NFc + 1) + c];
    else if (c < NFc + EFc) x = ef[(size_t)e * (EFc + 1) + (c - NFc)];
    Xe[i] = f2bf(x);
}

// h = relu(h0 + accZ[src[e]] - Z[e^1] + bm)
__global__ void k_combine(const bf16* __restrict__ h0, const float* __restrict__ accZ,
                          const bf16* __restrict__ Z, const float* __restrict__ bm,
                          const int* __restrict__ src, bf16* __restrict__ h) {
    int i = blockIdx.x * blockDim.x + threadIdx.x;
    if (i >= Ec * HP) return;
    int e = i / HP, c = i % HP;
    float b = (c < Hc) ? bm[c] : 0.f;
    float val = bf2f(h0[i]) + accZ[(size_t)src[e] * HP + c] - bf2f(Z[(size_t)(e ^ 1) * HP + c]) + b;
    h[i] = f2bf(val > 0.f ? val : 0.f);
}

__global__ void k_scatter_h(const bf16* __restrict__ h, const int* __restrict__ src,
                            float* __restrict__ mv) {
    int i = blockIdx.x * blockDim.x + threadIdx.x;
    if (i >= Ec * HP) return;
    int e = i / HP, c = i % HP;
    atomicAdd(mv + (size_t)src[e] * HP + c, bf2f(h[i]));
}

__global__ void k_build_xa(const float* __restrict__ v, const float* __restrict__ mv,
                           bf16* __restrict__ Xa) {
    int i = blockIdx.x * blockDim.x + threadIdx.x;
    if (i >= Nc * KAp) return;
    int n = i / KAp, c = i % KAp;
    float x = 0.f;
    if (c < NFc)           x = v[(size_t)n * (NFc + 1) + c];
    else if (c < NFc + Hc) x = mv[(size_t)n * HP + (c - NFc)];
    Xa[i] = f2bf(x);
}

__global__ void k_build_xr(const float* __restrict__ pooled, const float* __restrict__ g,
                           bf16* __restrict__ Xr) {
    int i = blockIdx.x * blockDim.x + threadIdx.x;
    if (i >= Bc * KRp) return;
    int b = i / KRp, c = i % KRp;
    float x = 0.f;
    if (c < Hc)           x = pooled[(size_t)b * HP + c] * (1.f / 32.f);
    else if (c < Hc + Gc) x = g[(size_t)b * Gc + (c - Hc)];
    Xr[i] = f2bf(x);
}

// ---------- LDS-staged double-buffered bf16 WMMA GEMM ----------
// Block = 256 threads = 8 waves; block tile 128(M) x 64(N), K-step 32.
// Wave (wm = w&1, wn = w>>1): 64 rows x 16 cols -> 4 WMMAs per K-step.
// Per K-step: ALL fragment loads (8 ds_load_b128 for A, 16 2-byte B loads) are issued first, then the
// async stage of the next tile, then the 4 WMMAs back-to-back (single dscnt
// drain instead of one wait per WMMA).
__global__ void __launch_bounds__(256)
k_gemm(const bf16* __restrict__ A, int lda,
       const bf16* __restrict__ B, int ldb,
       const float* __restrict__ bias, int nreal,
       int M, int Npad, int K, int relu,
       bf16* __restrict__ out0, bf16* __restrict__ out1,
       float* __restrict__ scat, const int* __restrict__ sidx,
       float* __restrict__ fout, int fld, int fnc)
{
    __shared__ bf16 sA[2][128][32];   // 2 x 8KB
    __shared__ bf16 sB[2][32][64];    // 2 x 4KB

    const int tid  = threadIdx.x;
    const int lane = tid & 31;
    const int w    = tid >> 5;
    const int wm   = w & 1;
    const int wn   = w >> 1;
    const int r    = lane & 15;
    const int hi   = lane >> 4;

    const int nBN  = Npad >> 6;
    const int bn   = blockIdx.x % nBN;
    const int bm   = blockIdx.x / nBN;
    const int rowB = bm << 7;
    const int colB = bn << 6;

    // staging work split: A = 512 x 16B segs (2/thread), B = 256 segs (1/thread)
    const int a0row = tid >> 2, aseg = tid & 3;
    const int a1row = a0row + 64;
    const int brow  = tid >> 3, bseg = tid & 7;

    const unsigned sAb[2] = { (unsigned)(uintptr_t)&sA[0][0][0],
                              (unsigned)(uintptr_t)&sA[1][0][0] };
    const unsigned sBb[2] = { (unsigned)(uintptr_t)&sB[0][0][0],
                              (unsigned)(uintptr_t)&sB[1][0][0] };

    auto stage = [&](int buf, int k0) {
        async_ld16(sAb[buf] + a0row * 64 + aseg * 16,
                   A + (size_t)(rowB + a0row) * lda + k0 + aseg * 8);
        async_ld16(sAb[buf] + a1row * 64 + aseg * 16,
                   A + (size_t)(rowB + a1row) * lda + k0 + aseg * 8);
        async_ld16(sBb[buf] + brow * 128 + bseg * 16,
                   B + (size_t)(k0 + brow) * ldb + colB + bseg * 8);
    };

    const v8f vz = {0.f, 0.f, 0.f, 0.f, 0.f, 0.f, 0.f, 0.f};
    v8f acc[4];
#pragma unroll
    for (int t = 0; t < 4; ++t) acc[t] = vz;

    stage(0, 0);
    wait_async0();
    __syncthreads();

    int buf = 0;
    for (int k0 = 0; k0 < K; k0 += 32) {
        const int nk = k0 + 32;

        // 1) issue ALL fragment loads from LDS (8 x ds_load_b128 for A, 16 x 2-byte for B)
        V16 bfr, afr[4];
        const bf16* bp = &sB[buf][hi * 8][wn * 16 + r];
#pragma unroll
        for (int s = 0; s < 16; ++s) bfr.v[s] = bp[((s & 7) + (s >> 3) * 16) * 64];
#pragma unroll
        for (int t = 0; t < 4; ++t) {
            const bf16* ap = &sA[buf][wm * 64 + t * 16 + r][hi * 8];
            afr[t].q[0] = *(const v4i*)ap;
            afr[t].q[1] = *(const v4i*)(ap + 16);
        }

        // 2) kick off the next tile's async copies while LDS loads are in flight
        if (nk < K) stage(buf ^ 1, nk);

        // 3) 4 WMMAs back-to-back
#pragma unroll
        for (int t = 0; t < 4; ++t)
            acc[t] = __builtin_amdgcn_wmma_f32_16x16x32_bf16(
                false, afr[t].v, false, bfr.v, (short)0, acc[t], false, false);

        if (nk < K) wait_async0();
        __syncthreads();
        buf ^= 1;
    }

    const int col = colB + wn * 16 + r;
    const float bcol = (bias && col < nreal) ? bias[col] : 0.f;
#pragma unroll
    for (int t = 0; t < 4; ++t) {
#pragma unroll
        for (int i = 0; i < 8; ++i) {
            int grow = rowB + wm * 64 + t * 16 + hi * 8 + i;
            float val = acc[t][i] + bcol;
            if (relu && val < 0.f) val = 0.f;
            size_t o = (size_t)grow * Npad + col;
            if (out0) out0[o] = f2bf(val);
            if (out1) out1[o] = f2bf(val);
            if (scat) atomicAdd(scat + (size_t)sidx[grow] * Npad + col, val);
            if (fout && col < fnc) fout[(size_t)grow * fld + col] = val;
        }
    }
}

// ---------- host ----------
static inline int cdiv(long a, long b) { return (int)((a + b - 1) / b); }

extern "C" void kernel_launch(void* const* d_in, const int* in_sizes, int n_in,
                              void* d_out, int out_size, void* d_ws, size_t ws_size,
                              hipStream_t stream) {
    (void)in_sizes; (void)n_in; (void)out_size; (void)ws_size;
    const float* v   = (const float*)d_in[0];
    const float* ef  = (const float*)d_in[1];
    const float* gl  = (const float*)d_in[2];
    const float* Wi  = (const float*)d_in[3];
    const float* bi  = (const float*)d_in[4];
    const float* Wm  = (const float*)d_in[5];
    const float* bm  = (const float*)d_in[6];
    const float* Wa  = (const float*)d_in[7];
    const float* ba  = (const float*)d_in[8];
    const float* Wr1 = (const float*)d_in[9];
    const float* br1 = (const float*)d_in[10];
    const float* Wr2 = (const float*)d_in[11];
    const float* br2 = (const float*)d_in[12];
    const float* Wr3 = (const float*)d_in[13];
    const float* br3 = (const float*)d_in[14];
    const int* src = (const int*)d_in[15];
    const int* dst = (const int*)d_in[16];
    const int* ng  = (const int*)d_in[17];
    float* out = (float*)d_out;

    char* ws = (char*)d_ws;
    size_t off = 0;
    auto alloc = [&](size_t bytes) -> void* {
        void* p = ws + off;
        off += (bytes + 255) & ~(size_t)255;
        return p;
    };
    bf16* WiB  = (bf16*)alloc((size_t)KIp * HP  * 2);
    bf16* WmB  = (bf16*)alloc((size_t)HP  * HP  * 2);
    bf16* WaB  = (bf16*)alloc((size_t)KAp * HP  * 2);
    bf16* Wr1B = (bf16*)alloc((size_t)KRp * R1p * 2);
    bf16* Wr2B = (bf16*)alloc((size_t)R1p * R2p * 2);
    bf16* Wr3B = (bf16*)alloc((size_t)R2p * TP  * 2);
    void* zxe  = alloc((size_t)Ec * HP * 2);        // Xe (E x KIp), later Z (E x HP)
    bf16* Xe   = (bf16*)zxe;
    bf16* Z    = (bf16*)zxe;
    bf16* h0   = (bf16*)alloc((size_t)Ec * HP * 2);
    bf16* h    = (bf16*)alloc((size_t)Ec * HP * 2);
    void* axa  = alloc((size_t)Nc * HP * 4);        // accZ (f32), later Xa (bf16)
    float* accZ = (float*)axa;
    bf16*  Xa   = (bf16*)axa;
    float* mv   = (float*)alloc((size_t)Nc * HP * 4);
    float* pooled = (float*)alloc((size_t)Bc * HP * 4);
    bf16* Xr = (bf16*)alloc((size_t)Bc * KRp * 2);
    bf16* r1 = (bf16*)alloc((size_t)Bc * R1p * 2);
    bf16* r2 = (bf16*)alloc((size_t)Bc * R2p * 2);

    const int TB = 256;
    auto gemm = [&](const bf16* A, int lda, const bf16* Bm, int ldb,
                    const float* bias, int nreal, int M, int Npad, int K, int relu,
                    bf16* o0, bf16* o1, float* sc, const int* si,
                    float* fo, int fld, int fnc) {
        int blocks = (M / 128) * (Npad / 64);
        k_gemm<<<blocks, TB, 0, stream>>>(A, lda, Bm, ldb, bias, nreal,
                                          M, Npad, K, relu, o0, o1, sc, si, fo, fld, fnc);
    };

    // weights -> padded bf16
    k_convw<<<cdiv((long)KIp * HP,  TB), TB, 0, stream>>>(Wi,  WiB,  NFc + EFc, Hc,  KIp, HP);
    k_convw<<<cdiv((long)HP  * HP,  TB), TB, 0, stream>>>(Wm,  WmB,  Hc,        Hc,  HP,  HP);
    k_convw<<<cdiv((long)KAp * HP,  TB), TB, 0, stream>>>(Wa,  WaB,  NFc + Hc,  Hc,  KAp, HP);
    k_convw<<<cdiv((long)KRp * R1p, TB), TB, 0, stream>>>(Wr1, Wr1B, Hc + Gc,   R1c, KRp, R1p);
    k_convw<<<cdiv((long)R1p * R2p, TB), TB, 0, stream>>>(Wr2, Wr2B, R1c,       R2c, R1p, R2p);
    k_convw<<<cdiv((long)R2p * TP,  TB), TB, 0, stream>>>(Wr3, Wr3B, R2c,       Tc,  R2p, TP);

    // h0 = relu([v[src], e] @ Wi + bi); seeds h
    k_build_xe<<<cdiv((long)Ec * KIp, TB), TB, 0, stream>>>(v, ef, src, Xe);
    gemm(Xe, KIp, WiB, HP, bi, Hc, Ec, HP, KIp, 1, h0, h, nullptr, nullptr, nullptr, 0, 0);

    // 3 layers: Z = h@Wm; accZ = segsum(Z,dst); h = relu(h0 + accZ[src] - Z[rev] + bm)
    for (int layer = 0; layer < 3; ++layer) {
        k_zero_f32<<<cdiv((long)Nc * HP, TB), TB, 0, stream>>>(accZ, Nc * HP);
        gemm(h, HP, WmB, HP, nullptr, 0, Ec, HP, HP, 0, Z, nullptr, accZ, dst, nullptr, 0, 0);
        k_combine<<<cdiv((long)Ec * HP, TB), TB, 0, stream>>>(h0, accZ, Z, bm, src, h);
    }

    // m_v = segsum(h, src); pooled = segsum(relu([v, m_v]@Wa + ba), node_graph)
    k_zero_f32<<<cdiv((long)Nc * HP, TB), TB, 0, stream>>>(mv, Nc * HP);
    k_scatter_h<<<cdiv((long)Ec * HP, TB), TB, 0, stream>>>(h, src, mv);
    k_build_xa<<<cdiv((long)Nc * KAp, TB), TB, 0, stream>>>(v, mv, Xa);
    k_zero_f32<<<cdiv((long)Bc * HP, TB), TB, 0, stream>>>(pooled, Bc * HP);
    gemm(Xa, KAp, WaB, HP, ba, Hc, Nc, HP, KAp, 1, nullptr, nullptr, pooled, ng, nullptr, 0, 0);

    // readout
    k_build_xr<<<cdiv((long)Bc * KRp, TB), TB, 0, stream>>>(pooled, gl, Xr);
    gemm(Xr, KRp, Wr1B, R1p, br1, R1c, Bc, R1p, KRp, 1, r1, nullptr, nullptr, nullptr, nullptr, 0, 0);
    gemm(r1, R1p, Wr2B, R2p, br2, R2c, Bc, R2p, R1p, 1, r2, nullptr, nullptr, nullptr, nullptr, 0, 0);
    gemm(r2, R2p, Wr3B, TP,  br3, Tc,  Bc, TP,  R2p, 0, nullptr, nullptr, nullptr, nullptr, out, Tc, Tc);
}
